// GNN_56633438765547
// MI455X (gfx1250) — matrix operand order repaired and checked by reading, not compiled, not run on hardware
//
#include <hip/hip_runtime.h>
#include <hip/hip_bf16.h>

typedef _Float16 v8h  __attribute__((ext_vector_type(8)));
typedef _Float16 v16h __attribute__((ext_vector_type(16)));
typedef float    v8f  __attribute__((ext_vector_type(8)));

#define NN    10000   // nodes
#define DD    256     // embedding dim
#define BB    8192    // pairs
#define HID   1025    // hidden width
#define K1    512     // 2*D
#define R2    16384   // 2*B gathered rows

// ---------------- Kernel 1: sparse gather-aggregate ----------------
// agg[r] = sum_{j: A[idx[r],j]!=0} A[idx[r],j]*E[j] + E[idx[r]]
// Each block = one gathered row, 8 waves scan disjoint column chunks of the
// A row (float4/lane, ballot-compacted), accumulate 256-dim partials in
// registers (8 f32/lane), reduce across waves in LDS, emit f16 activations.
__device__ __forceinline__ void acc_embed(float acc[8], const float* __restrict__ E,
                                          int j, float aj, int lane) {
  const float4* ep = (const float4*)(E + (size_t)j * DD + lane * 8);
  float4 e0 = ep[0];
  float4 e1 = ep[1];
  acc[0] += aj * e0.x; acc[1] += aj * e0.y; acc[2] += aj * e0.z; acc[3] += aj * e0.w;
  acc[4] += aj * e1.x; acc[5] += aj * e1.y; acc[6] += aj * e1.z; acc[7] += aj * e1.w;
}

__global__ void __launch_bounds__(256)
agg_kernel(const int* __restrict__ idx, const float* __restrict__ A,
           const float* __restrict__ E, _Float16* __restrict__ xout) {
  __shared__ float red[8][DD];
  const int r    = blockIdx.x;
  const int node = idx[r];
  const int wave = threadIdx.x >> 5;
  const int lane = threadIdx.x & 31;
  const float* Arow = A + (size_t)node * NN;

  float acc[8] = {0.f, 0.f, 0.f, 0.f, 0.f, 0.f, 0.f, 0.f};

  for (int c0 = wave * 128; c0 < NN; c0 += 1024) {
    const int c = c0 + lane * 4;
    float4 a4;
    if (c + 3 < NN) {
      a4 = *(const float4*)(Arow + c);
    } else {
      a4.x = (c + 0 < NN) ? Arow[c + 0] : 0.f;
      a4.y = (c + 1 < NN) ? Arow[c + 1] : 0.f;
      a4.z = (c + 2 < NN) ? Arow[c + 2] : 0.f;
      a4.w = (c + 3 < NN) ? Arow[c + 3] : 0.f;
    }
    const bool nz = (a4.x != 0.f) || (a4.y != 0.f) || (a4.z != 0.f) || (a4.w != 0.f);
    unsigned long long m = __ballot(nz);   // wave32: low 32 bits used
    while (m) {
      const int b = __ffsll(m) - 1;
      m &= m - 1;
      const float j0 = __shfl(a4.x, b);
      const float j1 = __shfl(a4.y, b);
      const float j2 = __shfl(a4.z, b);
      const float j3 = __shfl(a4.w, b);
      const int cb = c0 + b * 4;
      if (j0 != 0.f) acc_embed(acc, E, cb + 0, j0, lane);
      if (j1 != 0.f) acc_embed(acc, E, cb + 1, j1, lane);
      if (j2 != 0.f) acc_embed(acc, E, cb + 2, j2, lane);
      if (j3 != 0.f) acc_embed(acc, E, cb + 3, j3, lane);
    }
  }

  #pragma unroll
  for (int t = 0; t < 8; ++t) red[wave][lane * 8 + t] = acc[t];
  __syncthreads();

  const int d = threadIdx.x;
  float s = 0.f;
  #pragma unroll
  for (int w = 0; w < 8; ++w) s += red[w][d];
  s += E[(size_t)node * DD + d];          // identity row of (I + A)
  xout[(size_t)r * DD + d] = (_Float16)s; // f16 activations for WMMA stage
}

// ---------------- Kernel 2: h = relu(x @ W1 + b1) via WMMA f16 ----------------
// Block tile: 128 rows (8 waves x 16) x 64 cols (4 n-tiles/wave).
// W1 panel staged transposed into LDS as f16: Ws[n][k] so B-fragments are
// two 8-half k runs per lane (k0+8*hi and k0+16+8*hi), per the ISA B layout.
__global__ void __launch_bounds__(256)
mlp1_kernel(const _Float16* __restrict__ x, const float* __restrict__ W1,
            const float* __restrict__ b1, float* __restrict__ h) {
  __shared__ _Float16 Ws[64 * K1];   // 64 KB
  const int tid = threadIdx.x;
  const int n0  = blockIdx.y * 64;
  const int m0  = blockIdx.x * 128;

  for (int i = tid; i < 64 * K1; i += 256) {
    const int k   = i >> 6;        // 0..511
    const int j   = i & 63;        // 0..63
    const int col = n0 + j;
    const float v = (col < HID) ? W1[k * HID + col] : 0.f;  // zero-pad N edge
    Ws[j * K1 + k] = (_Float16)v;
  }
  __syncthreads();

  const int wave = tid >> 5;
  const int lane = tid & 31;
  const int hi   = lane >> 4;      // lane-half select
  const int lo   = lane & 15;
  const int mrow = m0 + wave * 16 + lo;
  const _Float16* xrow = x + (size_t)mrow * K1;

  v8f acc[4] = {};
  for (int k0 = 0; k0 < K1; k0 += 32) {
    // A 16x32 f16 layout: lane holds K = k0+8*hi..+7 and k0+16+8*hi..+7
    const v8h a0 = *(const v8h*)(xrow + k0 + 8 * hi);
    const v8h a1 = *(const v8h*)(xrow + k0 + 16 + 8 * hi);
    const v16h afrag = __builtin_shufflevector(a0, a1,
        0, 1, 2, 3, 4, 5, 6, 7, 8, 9, 10, 11, 12, 13, 14, 15);
    #pragma unroll
    for (int nt = 0; nt < 4; ++nt) {
      // B 32x16 f16 layout: lane = column, K = k0+8*hi..+7 and k0+16+8*hi..+7
      const _Float16* bp = &Ws[(nt * 16 + lo) * K1 + k0 + 8 * hi];
      const v8h b0  = *(const v8h*)bp;
      const v8h b1v = *(const v8h*)(bp + 16);
      const v16h bfrag = __builtin_shufflevector(b0, b1v,
          0, 1, 2, 3, 4, 5, 6, 7, 8, 9, 10, 11, 12, 13, 14, 15);
      acc[nt] = __builtin_amdgcn_wmma_f32_16x16x32_f16(
          /*neg_a=*/false, afrag, /*neg_b=*/false, bfrag,
          /*c_mod=*/(short)0, acc[nt], /*reuse_a=*/false, /*reuse_b=*/false);
    }
  }

  #pragma unroll
  for (int nt = 0; nt < 4; ++nt) {
    const int col = n0 + nt * 16 + lo;
    if (col < HID) {
      const float bias = b1[col];
      #pragma unroll
      for (int i = 0; i < 8; ++i) {
        const int row = m0 + wave * 16 + i + 8 * hi;  // C/D layout: VGPR i -> M=i (+8 for hi half)
        float v = acc[nt][i] + bias;
        h[(size_t)row * HID + col] = fmaxf(v, 0.f);
      }
    }
  }
}

// ---------------- Kernel 3: out = h @ W2 + b2 (N=2) ----------------
__global__ void __launch_bounds__(256)
mlp2_kernel(const float* __restrict__ h, const float* __restrict__ W2,
            const float* __restrict__ b2, float* __restrict__ out) {
  const int row  = blockIdx.x * 8 + (threadIdx.x >> 5);
  const int lane = threadIdx.x & 31;
  const float* hrow = h + (size_t)row * HID;
  const float2* w2v = (const float2*)W2;
  float p0 = 0.f, p1 = 0.f;
  for (int c = lane; c < HID; c += 32) {
    const float  hv = hrow[c];
    const float2 w  = w2v[c];
    p0 += hv * w.x;
    p1 += hv * w.y;
  }
  #pragma unroll
  for (int off = 16; off > 0; off >>= 1) {
    p0 += __shfl_down(p0, off);
    p1 += __shfl_down(p1, off);
  }
  if (lane == 0) {
    out[row * 2 + 0] = p0 + b2[0];
    out[row * 2 + 1] = p1 + b2[1];
  }
}

extern "C" void kernel_launch(void* const* d_in, const int* in_sizes, int n_in,
                              void* d_out, int out_size, void* d_ws, size_t ws_size,
                              hipStream_t stream) {
  const int*   idx = (const int*)  d_in[0];  // [B,2] int32
  const float* A   = (const float*)d_in[1];  // [N,N]
  const float* E   = (const float*)d_in[2];  // [N,D]
  const float* W1  = (const float*)d_in[3];  // [2D,HID]
  const float* b1  = (const float*)d_in[4];  // [HID]
  const float* W2  = (const float*)d_in[5];  // [HID,2]
  const float* b2  = (const float*)d_in[6];  // [2]
  float* out = (float*)d_out;                // [B,2]

  // Workspace layout: x (f16, 16384x256 = 8 MB) | h (f32, 8192x1025 = 33.6 MB)
  _Float16* x = (_Float16*)d_ws;
  float*    h = (float*)((char*)d_ws + (size_t)R2 * DD * sizeof(_Float16));

  agg_kernel<<<R2, 256, 0, stream>>>(idx, A, E, x);
  mlp1_kernel<<<dim3(BB / 128, (HID + 63) / 64), 256, 0, stream>>>(x, W1, b1, h);
  mlp2_kernel<<<BB / 8, 256, 0, stream>>>(h, W2, b2, out);
}
